// Generator_26714696581262
// MI455X (gfx1250) — compile-verified, matrix operand order checked by reading, not run on hardware
//
#include <hip/hip_runtime.h>
#include <hip/hip_bf16.h>

// ---------------------------------------------------------------------------
// CDNA5 (gfx1250, wave32) graph U-Net with GAT layers.
// GEMMs: v_wmma_f32_16x16x32_bf16, tiles staged via Tensor Data Mover (TDM)
// when available (tensor_load_to_lds + s_wait_tensorcnt), else b128 staging.
// Edge softmax: f32/u32 global atomics.
// ---------------------------------------------------------------------------

typedef __attribute__((ext_vector_type(16))) __bf16    v16bf;
typedef __attribute__((ext_vector_type(8)))  float     v8f;
typedef __attribute__((ext_vector_type(4)))  unsigned  v4u;
typedef __attribute__((ext_vector_type(8)))  int       v8i;
typedef __attribute__((ext_vector_type(4)))  int       v4i;

#define NEG_SLOPE 0.2f

#if defined(__has_builtin)
#  if __has_builtin(__builtin_amdgcn_tensor_load_to_lds)
#    define USE_TDM 1
#  endif
#endif

__device__ __forceinline__ unsigned f2ord(float f) {
  unsigned u = __float_as_uint(f);
  return (u & 0x80000000u) ? ~u : (u | 0x80000000u);   // monotonic float->uint
}
__device__ __forceinline__ float ord2f(unsigned u) {
  return __uint_as_float((u & 0x80000000u) ? (u & 0x7fffffffu) : ~u);
}

#ifdef USE_TDM
// Issue one TDM 2D tile load: global (row-major, elem=f32, row pitch `stride0`
// elements) -> LDS at byte offset lds_off, with per-row padding so the LDS row
// stride becomes (tile_d0 + pad) dwords. D# layout per CDNA5 ISA ch.8.
// This toolchain's builtin is the 6-arg form:
//   (uint32x4 group0, int32x8 group1, int32x4, int32x4, int32x8, i32 cpol)
__device__ __forceinline__ void tdm_load_tile_f32(
    const float* gaddr, unsigned lds_off,
    unsigned tile_d0, unsigned tile_d1,
    unsigned long long tensor_d0, unsigned long long tensor_d1,
    unsigned long long stride0,
    unsigned pad_interval_code, unsigned pad_amount_code)
{
  unsigned long long ga = (unsigned long long)(size_t)gaddr;

  // group0: count=1 | lds_addr<<32 ; global_addr[56:0] | type(2)<<62
  unsigned long long g0q0 = 1ull | ((unsigned long long)lds_off << 32);
  unsigned long long g0q1 = (ga & 0x01FFFFFFFFFFFFFFull) | (2ull << 62);

  // group1 word0: data_size=2 (4B) | pad_enable | pad_interval | pad_amount
  unsigned w0 = (2u << 16) | (1u << 20)
              | (pad_interval_code << 22) | (pad_amount_code << 25);
  unsigned long long q0 = (unsigned long long)w0
                        | ((tensor_d0 & 0xFFFFull) << 48);          // td0[15:0]
  unsigned long long q1 = ((tensor_d0 >> 16) & 0xFFFFull)           // td0[31:16]
                        | ((tensor_d1 & 0xFFFFFFFFull) << 16)       // td1
                        | ((unsigned long long)(tile_d0 & 0xFFFFu) << 48);
  unsigned long long q2 = (unsigned long long)(tile_d1 & 0xFFFFu)   // tile_d1
                        | ((stride0 & 0xFFFFFFFFull) << 32);        // s0[31:0]
  unsigned long long q3 = (stride0 >> 32) & 0xFFFFull;              // s0[47:32]

  v4u g0 = { (unsigned)g0q0, (unsigned)(g0q0 >> 32),
             (unsigned)g0q1, (unsigned)(g0q1 >> 32) };
  v8i g1 = { (int)q0, (int)(q0 >> 32), (int)q1, (int)(q1 >> 32),
             (int)q2, (int)(q2 >> 32), (int)q3, (int)(q3 >> 32) };
  v4i z4 = { 0, 0, 0, 0 };
  v8i z8 = { 0, 0, 0, 0, 0, 0, 0, 0 };
  __builtin_amdgcn_tensor_load_to_lds(g0, g1, z4, z4, z8, 0);
}
#endif

// ---------------------------------------------------------------------------
// C[n,Cout] = A[n,K] @ W[K,Cout]   (bf16 WMMA, f32 accumulate)
// block = 128 threads = 4 waves; block tile 64x64; wave tile 32x32
// (= 4 x 16x16 WMMA tiles per wave per K step); K step 32.
// K mult of 32, Cout mult of 64 for every layer in this model.
// LDS: A tile 64x32 @ row stride 36 dwords (offset 0),
//      B tile 32x64 @ row stride 68 dwords (offset 9216 B).
// ---------------------------------------------------------------------------
#define ALS(r, c) Lv[(r) * 36 + (c)]
#define BLS(r, c) Lv[2304 + (r) * 68 + (c)]

__global__ __launch_bounds__(128) void gemm_xw_wmma(
    const float* __restrict__ A, const float* __restrict__ W,
    float* __restrict__ C, int n, int K, int Cout)
{
  __shared__ __align__(16) float lds[64 * 36 + 32 * 68];

  const int tid  = threadIdx.x;
  const int lane = tid & 31;
  const int wave = tid >> 5;
  const int wm   = (wave >> 1) << 5;   // wave row offset inside 64x64 tile
  const int wn   = (wave & 1) << 5;    // wave col offset
  const int m0   = blockIdx.x << 6;
  const int n0   = blockIdx.y << 6;
  const int half = lane >> 4;          // 0: lanes 0-15, 1: lanes 16-31
  const int lr   = lane & 15;

  // The TDM engine writes `lds` out-of-band (invisible to alias analysis).
  // Escape the address so memory clobbers cover it, and read fragments
  // through a volatile pointer so the gathers can never fold to undef.
  asm volatile("" :: "v"(&lds[0]) : "memory");
  volatile const float* Lv = lds;

  v8f acc00 = {}, acc01 = {}, acc10 = {}, acc11 = {};

  for (int k0 = 0; k0 < K; k0 += 32) {
#ifdef USE_TDM
    if (wave == 0) {
      // A tile: rows m0..m0+63 (zero-filled past n via tensor_dim1), K window 32.
      tdm_load_tile_f32(A + (size_t)m0 * K + k0, /*lds_off=*/0,
                        /*tile_d0=*/32, /*tile_d1=*/64,
                        /*tensor_d0=*/(unsigned long long)K,
                        /*tensor_d1=*/(unsigned long long)(n - m0),
                        /*stride0=*/(unsigned long long)K,
                        /*pad_interval(32dw)=*/4u, /*pad_amount(4dw)=*/3u);
      // B tile: K rows k0..k0+31, cols n0..n0+63 (always in-bounds).
      tdm_load_tile_f32(W + (size_t)k0 * Cout + n0, /*lds_off=*/9216u,
                        /*tile_d0=*/64, /*tile_d1=*/32,
                        /*tensor_d0=*/(unsigned long long)Cout,
                        /*tensor_d1=*/32ull,
                        /*stride0=*/(unsigned long long)Cout,
                        /*pad_interval(64dw)=*/5u, /*pad_amount(4dw)=*/3u);
      __builtin_amdgcn_s_wait_tensorcnt(0);
    }
    asm volatile("" ::: "memory");
    __syncthreads();
#else
    if (k0 + 32 < K) {                 // global_prefetch_b8 for next K tile
      __builtin_prefetch(A + (size_t)(m0 + lane) * K + k0 + 32, 0, 1);
      __builtin_prefetch(W + (size_t)(k0 + 32 + lane) * Cout + n0, 0, 1);
    }
    // A: 64x32 = 512 float4, B: 32x64 = 512 float4, 128 threads -> 4 each.
    for (int t = tid; t < 512; t += 128) {
      int ra = t >> 3, ca = (t & 7) << 2;          // A
      float4 av = make_float4(0.f, 0.f, 0.f, 0.f);
      int gm = m0 + ra;
      if (gm < n) av = *(const float4*)(A + (size_t)gm * K + k0 + ca);
      *(float4*)&lds[ra * 36 + ca] = av;
      int rb = t >> 4, cb = (t & 15) << 2;         // B
      float4 bv = *(const float4*)(W + (size_t)(k0 + rb) * Cout + n0 + cb);
      *(float4*)&lds[2304 + rb * 68 + cb] = bv;
    }
    __syncthreads();
#endif

    // Fragment gather per ISA 7.12.2 16-bit A 16x32 layout:
    //   VGPR v, pair p: K = (v<4 ? 2v : 16+2(v-4)) + 8*half + p
    v16bf a0, a1, b0, b1;
#pragma unroll
    for (int i = 0; i < 16; ++i) {
      int v = i >> 1, p = i & 1;
      int kk = ((v < 4) ? (v << 1) : (16 + ((v - 4) << 1))) + (half << 3) + p;
      a0[i] = (__bf16)ALS(wm + lr, kk);
      a1[i] = (__bf16)ALS(wm + 16 + lr, kk);
      b0[i] = (__bf16)BLS(kk, wn + lr);
      b1[i] = (__bf16)BLS(kk, wn + 16 + lr);
    }
    acc00 = __builtin_amdgcn_wmma_f32_16x16x32_bf16(false, a0, false, b0,
                                                    (short)0, acc00, false, false);
    acc01 = __builtin_amdgcn_wmma_f32_16x16x32_bf16(false, a0, false, b1,
                                                    (short)0, acc01, false, false);
    acc10 = __builtin_amdgcn_wmma_f32_16x16x32_bf16(false, a1, false, b0,
                                                    (short)0, acc10, false, false);
    acc11 = __builtin_amdgcn_wmma_f32_16x16x32_bf16(false, a1, false, b1,
                                                    (short)0, acc11, false, false);
    __syncthreads();
    // Keep LDS reads above ordered before next iteration's TDM overwrite.
    asm volatile("" ::: "memory");
  }

  // C/D layout: VGPR r -> M = r + 8*half, N = lane%16.
  const int gn = n0 + wn + lr;
#pragma unroll
  for (int r = 0; r < 8; ++r) {
    int gm = m0 + wm + (half << 3) + r;
    if (gm < n) {
      C[(size_t)gm * Cout + gn]      = acc00[r];
      C[(size_t)gm * Cout + gn + 16] = acc01[r];
    }
    int gm2 = gm + 16;
    if (gm2 < n) {
      C[(size_t)gm2 * Cout + gn]      = acc10[r];
      C[(size_t)gm2 * Cout + gn + 16] = acc11[r];
    }
  }
}

// ---------------------------------------------------------------------------
// Per-node attention logits: as[i,h] = sum_c h[i,h,c]*a_src[h,c] (same for ad)
// ---------------------------------------------------------------------------
__global__ void attn_ab(const float* __restrict__ h,
                        const float* __restrict__ a_src,
                        const float* __restrict__ a_dst,
                        float* __restrict__ as, float* __restrict__ ad,
                        int n, int H, int Cph)
{
  int gid = blockIdx.x * blockDim.x + threadIdx.x;
  if (gid >= n * H) return;
  int node = gid / H, hh = gid - node * H;
  const float* hp = h + (size_t)node * H * Cph + (size_t)hh * Cph;
  float s1 = 0.f, s2 = 0.f;
  for (int c = 0; c < Cph; ++c) {
    float v = hp[c];
    s1 += v * a_src[hh * Cph + c];
    s2 += v * a_dst[hh * Cph + c];
  }
  as[gid] = s1;
  ad[gid] = s2;
}

// ---------------------------------------------------------------------------
// Edge softmax pass 1: alpha = leaky_relu(as[src]+ad[dst]); atomic max per dst
// Edges e < E come from arrays; e in [E, E+n) are self loops (always valid).
// ---------------------------------------------------------------------------
__global__ void edge_alpha_max(const int* __restrict__ src,
                               const int* __restrict__ dst,
                               const unsigned char* __restrict__ mask,
                               const float* __restrict__ as,
                               const float* __restrict__ ad,
                               float* __restrict__ alpha,
                               unsigned* __restrict__ amax,
                               int E_, int n, int H)
{
  int gid = blockIdx.x * blockDim.x + threadIdx.x;
  if (gid >= (E_ + n) * H) return;
  int e = gid / H, hh = gid - e * H;
  int s, d; bool ok;
  if (e < E_) { s = src[e]; d = dst[e]; ok = mask[e] != 0; }
  else        { s = d = e - E_;         ok = true; }
  if (!ok) return;
  float a = as[s * H + hh] + ad[d * H + hh];
  a = (a > 0.f) ? a : NEG_SLOPE * a;
  alpha[gid] = a;
  __hip_atomic_fetch_max(&amax[d * H + hh], f2ord(a),
                         __ATOMIC_RELAXED, __HIP_MEMORY_SCOPE_AGENT);
}

// Pass 2: e = exp(alpha - amax[dst]); atomic add denom[dst]. In-place on ebuf.
__global__ void edge_exp_sum(const int* __restrict__ dst,
                             const unsigned char* __restrict__ mask,
                             float* __restrict__ ebuf,
                             const unsigned* __restrict__ amax,
                             float* __restrict__ denom,
                             int E_, int n, int H)
{
  int gid = blockIdx.x * blockDim.x + threadIdx.x;
  if (gid >= (E_ + n) * H) return;
  int e = gid / H, hh = gid - e * H;
  int d; bool ok;
  if (e < E_) { d = dst[e]; ok = mask[e] != 0; }
  else        { d = e - E_; ok = true; }
  if (!ok) { ebuf[gid] = 0.f; return; }
  float m  = ord2f(amax[d * H + hh]);
  float ev = expf(ebuf[gid] - m);
  ebuf[gid] = ev;
  __hip_atomic_fetch_add(&denom[d * H + hh], ev,
                         __ATOMIC_RELAXED, __HIP_MEMORY_SCOPE_AGENT);
}

// Pass 3: out[dst, f] += (e/denom[dst]) * h[src, f]   (thread per edge-feature)
__global__ void edge_aggregate(const int* __restrict__ src,
                               const int* __restrict__ dst,
                               const float* __restrict__ ebuf,
                               const float* __restrict__ denom,
                               const float* __restrict__ h,
                               float* __restrict__ out,
                               int E_, int n, int H, int cout)
{
  size_t gid = (size_t)blockIdx.x * blockDim.x + threadIdx.x;
  if (gid >= (size_t)(E_ + n) * cout) return;
  int e = (int)(gid / cout);
  int f = (int)(gid - (size_t)e * cout);
  int Cph = cout / H;
  int hh = f / Cph;
  float ev = ebuf[(size_t)e * H + hh];
  if (ev == 0.f) return;
  int s, d;
  if (e < E_) { s = src[e]; d = dst[e]; }
  else        { s = d = e - E_; }
  float den = denom[d * H + hh];
  den = den > 1e-16f ? den : 1e-16f;
  __hip_atomic_fetch_add(&out[(size_t)d * cout + f],
                         (ev / den) * h[(size_t)s * cout + f],
                         __ATOMIC_RELAXED, __HIP_MEMORY_SCOPE_AGENT);
}

__global__ void bias_act(float* __restrict__ x, const float* __restrict__ b,
                         int n, int c, int elu)
{
  size_t gid = (size_t)blockIdx.x * blockDim.x + threadIdx.x;
  if (gid >= (size_t)n * c) return;
  int f = (int)(gid % c);
  float v = x[gid] + b[f];
  if (elu) v = (v > 0.f) ? v : (expf(v) - 1.f);
  x[gid] = v;
}

__global__ void fill_u32(unsigned* __restrict__ p, unsigned v, size_t count)
{
  size_t gid = (size_t)blockIdx.x * blockDim.x + threadIdx.x;
  if (gid < count) p[gid] = v;
}

// ---------------------------------------------------------------------------
// TopK pooling
// ---------------------------------------------------------------------------
__global__ void vec_norm(const float* __restrict__ w, int c, float* __restrict__ out)
{
  __shared__ float red[256];
  float s = 0.f;
  for (int i = threadIdx.x; i < c; i += blockDim.x) { float v = w[i]; s += v * v; }
  red[threadIdx.x] = s;
  __syncthreads();
  for (int o = 128; o > 0; o >>= 1) {
    if ((int)threadIdx.x < o) red[threadIdx.x] += red[threadIdx.x + o];
    __syncthreads();
  }
  if (threadIdx.x == 0) out[0] = sqrtf(red[0]);
}

__global__ void score_kernel(const float* __restrict__ h, const float* __restrict__ w,
                             const float* __restrict__ norm,
                             float* __restrict__ score, int n, int c)
{
  int i = blockIdx.x * blockDim.x + threadIdx.x;
  if (i >= n) return;
  const float* hp = h + (size_t)i * c;
  float s = 0.f;
  for (int j = 0; j < c; ++j) s += hp[j] * w[j];
  score[i] = tanhf(s / norm[0]);
}

// Exact selection rank: #(s_j > s_i) + #(j<i with s_j == s_i). O(n^2), LDS-tiled.
__global__ void rank_kernel(const float* __restrict__ score, int* __restrict__ rank, int n)
{
  __shared__ float tile[256];
  int i = blockIdx.x * 256 + threadIdx.x;
  float si = (i < n) ? score[i] : 0.f;
  int r = 0;
  for (int base = 0; base < n; base += 256) {
    int j = base + threadIdx.x;
    tile[threadIdx.x] = (j < n) ? score[j] : -3.4e38f;
    __syncthreads();
    int lim = n - base; if (lim > 256) lim = 256;
    for (int t = 0; t < lim; ++t) {
      float sj = tile[t];
      int j2 = base + t;
      r += (sj > si) || (sj == si && j2 < i);
    }
    __syncthreads();
  }
  if (i < n) rank[i] = r;
}

__global__ void build_nodemap(const int* __restrict__ rank, int* __restrict__ nm,
                              int n, int k)
{
  int i = blockIdx.x * blockDim.x + threadIdx.x;
  if (i >= n) return;
  int r = rank[i];
  nm[i] = (r < k) ? r : -1;
}

// xn[rank, :] = h[i, :] * score[i]  for kept nodes
__global__ void pool_features(const float* __restrict__ h, const float* __restrict__ score,
                              const int* __restrict__ rank, float* __restrict__ xn,
                              int n, int k, int c)
{
  size_t gid = (size_t)blockIdx.x * blockDim.x + threadIdx.x;
  if (gid >= (size_t)n * c) return;
  int i = (int)(gid / c), f = (int)(gid - (size_t)i * c);
  int r = rank[i];
  if (r >= k) return;
  xn[(size_t)r * c + f] = h[gid] * score[i];
}

__global__ void remap_edges(const int* __restrict__ src, const int* __restrict__ dst,
                            const unsigned char* __restrict__ mask,
                            const int* __restrict__ nm,
                            int* __restrict__ nsrc, int* __restrict__ ndst,
                            unsigned char* __restrict__ nmask, int E_)
{
  int e = blockIdx.x * blockDim.x + threadIdx.x;
  if (e >= E_) return;
  int s = nm[src[e]], d = nm[dst[e]];
  bool ok = (mask[e] != 0) && s >= 0 && d >= 0;
  nsrc[e]  = ok ? s : 0;
  ndst[e]  = ok ? d : 0;
  nmask[e] = ok ? 1 : 0;
}

// z[i, 0:c1] = cur[i % cn, :]; z[i, c1:c1+c2] = sx[i, :]   (tile-upsample+concat)
__global__ void upsample_concat(const float* __restrict__ cur, int cn, int c1,
                                const float* __restrict__ sx, int c2,
                                float* __restrict__ z, int n)
{
  size_t gid = (size_t)blockIdx.x * blockDim.x + threadIdx.x;
  int ct = c1 + c2;
  if (gid >= (size_t)n * ct) return;
  int i = (int)(gid / ct);
  int f = (int)(gid - (size_t)i * ct);
  z[gid] = (f < c1) ? cur[(size_t)(i % cn) * c1 + f]
                    : sx[(size_t)i * c2 + (f - c1)];
}

// ---------------------------------------------------------------------------
// Host orchestration
// ---------------------------------------------------------------------------
extern "C" void kernel_launch(void* const* d_in, const int* in_sizes, int n_in,
                              void* d_out, int out_size, void* d_ws, size_t ws_size,
                              hipStream_t stream)
{
  (void)in_sizes; (void)n_in; (void)out_size; (void)ws_size;
  const int N0 = 30000, E = 480000, CIN = 128, H = 4;
  const int n1 = 24000, n2 = 14400, n3 = 5760;   // ceil(ratio * n) per level

  const float* x0 = (const float*)d_in[0];
  const int*   ei = (const int*)d_in[1];
  const int* src0 = ei;
  const int* dst0 = ei + E;

  struct GatP { const float *W, *as, *ad, *b; };
  int pi = 2;
  auto getp = [&]() -> GatP {
    GatP g;
    g.W  = (const float*)d_in[pi++];
    g.as = (const float*)d_in[pi++];
    g.ad = (const float*)d_in[pi++];
    g.b  = (const float*)d_in[pi++];
    return g;
  };
  GatP enc[3]; for (int i = 0; i < 3; ++i) enc[i] = getp();
  const float* pool[3]; for (int i = 0; i < 3; ++i) pool[i] = (const float*)d_in[pi++];
  GatP bot = getp();
  GatP dec[3]; for (int i = 0; i < 3; ++i) dec[i] = getp();
  GatP outp = getp();

  // bump allocator over d_ws
  char* base = (char*)d_ws;
  size_t off = 0;
  auto alloc = [&](size_t bytes) -> void* {
    void* p = base + off;
    off = (off + bytes + 255) & ~(size_t)255;
    return p;
  };

  unsigned char* mask0 = (unsigned char*)alloc(E);
  const int *srcv[4], *dstv[4]; const unsigned char* maskv[4];
  srcv[0] = src0; dstv[0] = dst0; maskv[0] = mask0;
  int* srcA[3]; int* dstA[3]; unsigned char* maskA[3];
  for (int l = 0; l < 3; ++l) {
    srcA[l]  = (int*)alloc((size_t)E * 4);
    dstA[l]  = (int*)alloc((size_t)E * 4);
    maskA[l] = (unsigned char*)alloc(E);
    srcv[l + 1] = srcA[l]; dstv[l + 1] = dstA[l]; maskv[l + 1] = maskA[l];
  }

  float* x1 = (float*)alloc((size_t)n1 * CIN * 4);
  float* x2 = (float*)alloc((size_t)n2 * CIN * 4);
  float* x3 = (float*)alloc((size_t)n3 * CIN * 4);

  float*    hbuf  = (float*)alloc((size_t)N0 * 256 * 4);
  float*    asb   = (float*)alloc((size_t)N0 * H * 4);
  float*    adb   = (float*)alloc((size_t)N0 * H * 4);
  unsigned* amaxb = (unsigned*)alloc((size_t)N0 * H * 4);
  float*    denb  = (float*)alloc((size_t)N0 * H * 4);
  float*    ebuf  = (float*)alloc((size_t)(E + N0) * H * 4);

  float* g_enc0  = (float*)alloc((size_t)N0 * 128 * 4);
  float* g_enc1  = (float*)alloc((size_t)n1 * 128 * 4);
  float* g_enc2  = (float*)alloc((size_t)n2 * 128 * 4);
  float* bot_out = (float*)alloc((size_t)n3 * 256 * 4);
  float* cat0    = (float*)alloc((size_t)n2 * 384 * 4);
  float* d0o     = (float*)alloc((size_t)n2 * 128 * 4);
  float* cat1    = (float*)alloc((size_t)n1 * 256 * 4);
  float* d1o     = (float*)alloc((size_t)n1 * 128 * 4);
  float* cat2    = (float*)alloc((size_t)N0 * 256 * 4);
  float* d2o     = (float*)alloc((size_t)N0 * 128 * 4);

  float* scoreb = (float*)alloc((size_t)N0 * 4);
  int*   rankb  = (int*)alloc((size_t)N0 * 4);
  int*   nmapb  = (int*)alloc((size_t)N0 * 4);
  float* normb  = (float*)alloc(256);

  auto fill = [&](void* p, unsigned v, size_t words) {
    fill_u32<<<(unsigned)((words + 255) / 256), 256, 0, stream>>>((unsigned*)p, v, words);
  };

  auto run_gat = [&](const float* x, int n, int cin,
                     const int* src, const int* dst, const unsigned char* mask,
                     const GatP& P, int Hh, int cout, int elu, float* out) {
    dim3 gg((n + 63) / 64, cout / 64);
    gemm_xw_wmma<<<gg, 128, 0, stream>>>(x, P.W, hbuf, n, cin, cout);
    int Cph = cout / Hh;
    int nh = n * Hh;
    attn_ab<<<(nh + 255) / 256, 256, 0, stream>>>(hbuf, P.as, P.ad, asb, adb, n, Hh, Cph);
    fill(amaxb, 0u, (size_t)nh);
    fill(denb,  0u, (size_t)nh);
    fill(out,   0u, (size_t)n * cout);
    int t1 = (E + n) * Hh;
    edge_alpha_max<<<(t1 + 255) / 256, 256, 0, stream>>>(src, dst, mask, asb, adb,
                                                         ebuf, amaxb, E, n, Hh);
    edge_exp_sum<<<(t1 + 255) / 256, 256, 0, stream>>>(dst, mask, ebuf, amaxb, denb,
                                                       E, n, Hh);
    size_t t3 = (size_t)(E + n) * cout;
    edge_aggregate<<<(unsigned)((t3 + 255) / 256), 256, 0, stream>>>(src, dst, ebuf, denb,
                                                                     hbuf, out, E, n, Hh, cout);
    bias_act<<<(unsigned)(((size_t)n * cout + 255) / 256), 256, 0, stream>>>(out, P.b, n, cout, elu);
  };

  auto run_pool = [&](const float* h, int n, int c, const float* w, int k,
                      const int* src, const int* dst, const unsigned char* mask,
                      float* xn, int* nsrc, int* ndst, unsigned char* nmask) {
    vec_norm<<<1, 256, 0, stream>>>(w, c, normb);
    score_kernel<<<(n + 255) / 256, 256, 0, stream>>>(h, w, normb, scoreb, n, c);
    rank_kernel<<<(n + 255) / 256, 256, 0, stream>>>(scoreb, rankb, n);
    pool_features<<<(unsigned)(((size_t)n * c + 255) / 256), 256, 0, stream>>>(
        h, scoreb, rankb, xn, n, k, c);
    build_nodemap<<<(n + 255) / 256, 256, 0, stream>>>(rankb, nmapb, n, k);
    remap_edges<<<(E + 255) / 256, 256, 0, stream>>>(src, dst, mask, nmapb,
                                                     nsrc, ndst, nmask, E);
  };

  // level-0 edge mask = all true
  fill(mask0, 0x01010101u, E / 4);

  // --- encoder (down path) ---
  run_gat(x0, N0, CIN, srcv[0], dstv[0], maskv[0], enc[0], H, 128, 1, g_enc0);
  run_pool(g_enc0, N0, 128, pool[0], n1, srcv[0], dstv[0], maskv[0],
           x1, srcA[0], dstA[0], maskA[0]);
  run_gat(x1, n1, 128, srcv[1], dstv[1], maskv[1], enc[1], H, 128, 1, g_enc1);
  run_pool(g_enc1, n1, 128, pool[1], n2, srcv[1], dstv[1], maskv[1],
           x2, srcA[1], dstA[1], maskA[1]);
  run_gat(x2, n2, 128, srcv[2], dstv[2], maskv[2], enc[2], H, 128, 1, g_enc2);
  run_pool(g_enc2, n2, 128, pool[2], n3, srcv[2], dstv[2], maskv[2],
           x3, srcA[2], dstA[2], maskA[2]);

  // --- bottleneck ---
  run_gat(x3, n3, 128, srcv[3], dstv[3], maskv[3], bot, H, 256, 1, bot_out);

  // --- decoder (up path): tile-upsample + skip concat, then GAT ---
  {
    size_t t = (size_t)n2 * 384;
    upsample_concat<<<(unsigned)((t + 255) / 256), 256, 0, stream>>>(
        bot_out, n3, 256, x2, 128, cat0, n2);
  }
  run_gat(cat0, n2, 384, srcv[2], dstv[2], maskv[2], dec[0], H, 128, 1, d0o);
  {
    size_t t = (size_t)n1 * 256;
    upsample_concat<<<(unsigned)((t + 255) / 256), 256, 0, stream>>>(
        d0o, n2, 128, x1, 128, cat1, n1);
  }
  run_gat(cat1, n1, 256, srcv[1], dstv[1], maskv[1], dec[1], H, 128, 1, d1o);
  {
    size_t t = (size_t)N0 * 256;
    upsample_concat<<<(unsigned)((t + 255) / 256), 256, 0, stream>>>(
        d1o, n1, 128, x0, 128, cat2, N0);
  }
  run_gat(cat2, N0, 256, srcv[0], dstv[0], maskv[0], dec[2], H, 128, 1, d2o);

  // --- output layer: single head, mean over 1 head == identity, no ELU ---
  run_gat(d2o, N0, 128, srcv[0], dstv[0], maskv[0], outp, 1, 128, 0, (float*)d_out);
}
